// CnnLstmCrf_33947421508247
// MI455X (gfx1250) — matrix operand order repaired and checked by reading, not compiled, not run on hardware
//
#include <hip/hip_runtime.h>
#include <hip/hip_bf16.h>

// ---------------- WMMA bf16 helpers (gfx1250, wave32) ----------------
typedef __attribute__((ext_vector_type(16))) __bf16      v16bf;
typedef __attribute__((ext_vector_type(8)))  float       v8f;
typedef __attribute__((ext_vector_type(4)))  unsigned int u32x4;

union FragU { u32x4 q[2]; v16bf v; };

__device__ __forceinline__ unsigned short f2bf(float f) {
  unsigned int x = __float_as_uint(f);
  unsigned int r = x + 0x7FFFu + ((x >> 16) & 1u);   // RNE
  return (unsigned short)(r >> 16);
}
__device__ __forceinline__ float bf2f(unsigned short h) {
  return __uint_as_float(((unsigned int)h) << 16);
}
__device__ __forceinline__ float sigf(float x) { return 1.0f / (1.0f + __expf(-x)); }

// A-matrix 16x32 bf16 fragment: lane holds row m=lane&15, half=lane>>4,
// elements 0..7 = K {8h..8h+7}, elements 8..15 = K {16+8h..16+8h+7}.
__device__ __forceinline__ v16bf load_frag_a(const unsigned short* base, int ldk, int lane) {
  int row = lane & 15, half = lane >> 4;
  const unsigned short* p0 = base + row * ldk + 8 * half;
  FragU f;
  f.q[0] = *(const u32x4*)p0;
  f.q[1] = *(const u32x4*)(p0 + 16);
  return f.v;
}
// B-matrix 32x16 bf16 fragment: lane holds col n=lane&15, half=lane>>4,
// elements 0..7 = K {8h..8h+7}, elements 8..15 = K {16+8h..16+8h+7}. Weight stored n-major, k-contig.
__device__ __forceinline__ v16bf load_frag_b(const unsigned short* base, int ldk, int lane) {
  int col = lane & 15, half = lane >> 4;
  const unsigned short* p = base + col * ldk + 8 * half;
  FragU f;
  f.q[0] = *(const u32x4*)p;
  f.q[1] = *(const u32x4*)(p + 16);
  return f.v;
}
__device__ __forceinline__ v8f wmma_bf16(v16bf a, v16bf b, v8f c) {
  return __builtin_amdgcn_wmma_f32_16x16x32_bf16(false, a, false, b, (short)0, c, false, false);
}
__device__ __forceinline__ v8f vzero8() { v8f v = {0,0,0,0,0,0,0,0}; return v; }

// ---------------- problem dims ----------------
#define BB 64
#define TT 256
#define LL 16
#define DCC 50
#define CHH 100
#define DWW 300
#define DFF 20
#define HH 256
#define NN 52      // LAB + 2 ; START=50 STOP=51
#define INP 448    // 420 padded to 448 (mult of 32)
#define G4H 1024   // 4*H

// ---------------- k_prep: fp32 weights -> padded bf16 (n-major,k-contig) ----
__global__ void k_prep(const float* wihf, const float* wihb, const float* whhf,
                       const float* whhb, const float* projw, const float* convw,
                       unsigned short* o_wihf, unsigned short* o_wihb,
                       unsigned short* o_whhf, unsigned short* o_whhb,
                       unsigned short* o_projw, unsigned short* o_convw) {
  int id0 = blockIdx.x * blockDim.x + threadIdx.x;
  int stride = gridDim.x * blockDim.x;
  const int NWIH = G4H * INP, NWHH = G4H * HH, NPROJ = 64 * 512, NCONV = 112 * 160;
  for (int id = id0; id < NWIH; id += stride) {
    int n = id / INP, k = id % INP;
    o_wihf[id] = f2bf(k < 420 ? wihf[n * 420 + k] : 0.f);
    o_wihb[id] = f2bf(k < 420 ? wihb[n * 420 + k] : 0.f);
  }
  for (int id = id0; id < NWHH; id += stride) {
    o_whhf[id] = f2bf(whhf[id]);
    o_whhb[id] = f2bf(whhb[id]);
  }
  for (int id = id0; id < NPROJ; id += stride) {
    int n = id / 512, k = id % 512;
    o_projw[id] = f2bf(n < NN ? projw[n * 512 + k] : 0.f);
  }
  for (int id = id0; id < NCONV; id += stride) {
    int n = id / 160, k = id % 160;   // n = out channel, k = tap*50+dc
    float v = 0.f;
    if (n < CHH && k < 150) { int tap = k / DCC, dc = k % DCC; v = convw[n * 150 + dc * 3 + tap]; }
    o_convw[id] = f2bf(v);
  }
}

// ---------------- k_charcnn: 1 wave per word, WMMA 16x112 conv-GEMM + maxpool
__global__ __launch_bounds__(128) void k_charcnn(
    const int* batch_char, const float* char_emb, const unsigned short* convw_bf,
    const float* conv_b, float* pooled) {
  __shared__ __align__(16) unsigned short im2col[4][LL * 160];
  int wave = threadIdx.x >> 5, lane = threadIdx.x & 31;
  int word = blockIdx.x * 4 + wave;
  const int* ch = batch_char + word * LL;
  for (int e = lane; e < LL * 160; e += 32) {
    int p = e / 160, k = e % 160;
    float v = 0.f;
    if (k < 150) {
      int tap = k / DCC, dc = k % DCC;
      int cpos = p + tap - 1;                      // SAME padding, width-3 conv
      if (cpos >= 0 && cpos < LL) v = char_emb[ch[cpos] * DCC + dc];
    }
    im2col[wave][p * 160 + k] = f2bf(v);
  }
  __syncthreads();
  v8f acc[7];
#pragma unroll
  for (int nt = 0; nt < 7; ++nt) acc[nt] = vzero8();
  for (int kt = 0; kt < 5; ++kt) {
    v16bf a = load_frag_a(&im2col[wave][kt * 32], 160, lane);
#pragma unroll
    for (int nt = 0; nt < 7; ++nt) {
      v16bf b = load_frag_b(convw_bf + nt * 16 * 160 + kt * 32, 160, lane);
      acc[nt] = wmma_bf16(a, b, acc[nt]);
    }
  }
#pragma unroll
  for (int nt = 0; nt < 7; ++nt) {
    float m = acc[nt][0];
#pragma unroll
    for (int r = 1; r < 8; ++r) m = fmaxf(m, acc[nt][r]);   // max over 8 C rows
    m = fmaxf(m, __shfl_xor(m, 16, 32));                    // combine lane halves
    int n = nt * 16 + (lane & 15);
    if (lane < 16 && n < CHH) pooled[word * CHH + n] = m + conv_b[n];
  }
}

// ---------------- k_buildx: gather embeddings -> bf16 X (16384 x 448) -------
__global__ void k_buildx(const int* batch_word, const int* batch_features,
                         const int* recover, const float* word_emb,
                         const float* feat_emb, const float* pooled,
                         unsigned short* x_bf) {
  long total = (long)BB * TT * INP;
  for (long id = (long)blockIdx.x * blockDim.x + threadIdx.x; id < total;
       id += (long)gridDim.x * blockDim.x) {
    int bt = (int)(id / INP), col = (int)(id % INP);
    float v = 0.f;
    if (col < DWW)            v = word_emb[(long)batch_word[bt] * DWW + col];
    else if (col < DWW + CHH) v = pooled[recover[bt] * CHH + (col - DWW)];
    else if (col < 420)       v = feat_emb[batch_features[bt / TT] * DFF + (col - DWW - CHH)];
    x_bf[id] = f2bf(v);
  }
}

// ---------------- k_inproj: xs = X @ Wih^T + b, both dirs, -> [t][b][1024] bf16
__global__ __launch_bounds__(256) void k_inproj(
    const unsigned short* x_bf, const unsigned short* wihf_bf,
    const unsigned short* wihb_bf, const float* b_f, const float* b_b,
    unsigned short* xs_f, unsigned short* xs_b) {
  int gw = blockIdx.x * 8 + (threadIdx.x >> 5);   // 32768 waves
  int lane = threadIdx.x & 31;
  int mt = gw >> 5;        // 1024 m-tiles of 16 rows (row = b*T+t)
  int strip = gw & 31;     // 32 strips of 64 cols over 2048 (fwd|bwd)
  v8f acc[4];
#pragma unroll
  for (int nt = 0; nt < 4; ++nt) acc[nt] = vzero8();
  const unsigned short* abase = x_bf + (long)mt * 16 * INP;
  for (int kt = 0; kt < INP / 32; ++kt) {
    v16bf a = load_frag_a(abase + kt * 32, INP, lane);
#pragma unroll
    for (int nt = 0; nt < 4; ++nt) {
      int cg = strip * 64 + nt * 16;
      const unsigned short* w = (cg >> 10) ? wihb_bf : wihf_bf;
      int cl = cg & 1023;
      v16bf b = load_frag_b(w + (long)cl * INP + kt * 32, INP, lane);
      acc[nt] = wmma_bf16(a, b, acc[nt]);
    }
  }
#pragma unroll
  for (int nt = 0; nt < 4; ++nt) {
    int cg = strip * 64 + nt * 16 + (lane & 15);
    int dir = cg >> 10, cl = cg & 1023;
    float bias = (dir ? b_b : b_f)[cl];
    unsigned short* xs = dir ? xs_b : xs_f;
#pragma unroll
    for (int r = 0; r < 8; ++r) {
      int m = mt * 16 + 8 * (lane >> 4) + r;   // m = b*T + t
      int bb = m >> 8, t = m & 255;
      xs[((long)(t * BB + bb) * G4H) + cl] = f2bf(acc[nt][r] + bias);
    }
  }
}

// ---------------- k_scan: masked LSTM scan, one workgroup per direction -----
// LDS: h ping-pong bf16 (2x32KB) + c f32 (64KB) = 128KB of the 320KB WGP LDS.
__global__ __launch_bounds__(1024) void k_scan(
    const unsigned short* xs_f, const unsigned short* xs_b,
    const unsigned short* whhf_bf, const unsigned short* whhb_bf,
    const int* wordlen, unsigned short* hs) {
  __shared__ __align__(16) unsigned short h_sm[2][BB * HH];
  __shared__ float c_sm[BB * HH];
  int dir = blockIdx.x;
  const unsigned short* xs  = dir ? xs_b : xs_f;
  const unsigned short* whh = dir ? whhb_bf : whhf_bf;
  int lane = threadIdx.x & 31, wave = threadIdx.x >> 5;
  for (int i = threadIdx.x; i < BB * HH; i += 1024) {
    h_sm[0][i] = 0; h_sm[1][i] = 0; c_sm[i] = 0.f;
  }
  __syncthreads();
  for (int step = 0; step < TT; ++step) {
    int t = dir ? (TT - 1 - step) : step;
    int p = step & 1;                       // read h_sm[p], write h_sm[p^1]
#pragma unroll
    for (int u = 0; u < 2; ++u) {           // 2 (mtile,jgroup) units per wave
      int unit = wave * 2 + u;
      int mt = unit >> 4, jg = unit & 15;
      v8f acc[4];
#pragma unroll
      for (int g = 0; g < 4; ++g) acc[g] = vzero8();
      for (int kt = 0; kt < HH / 32; ++kt) {
        v16bf a = load_frag_a(&h_sm[p][mt * 16 * HH + kt * 32], HH, lane);
#pragma unroll
        for (int g = 0; g < 4; ++g) {       // gate n-tiles i,f,g,o at g*256+jg*16
          v16bf b = load_frag_b(whh + (long)(g * HH + jg * 16) * HH + kt * 32, HH, lane);
          acc[g] = wmma_bf16(a, b, acc[g]);
        }
      }
      int jc = jg * 16 + (lane & 15);
#pragma unroll
      for (int r = 0; r < 8; ++r) {
        int m = mt * 16 + 8 * (lane >> 4) + r;      // m = batch index
        const unsigned short* xrow = xs + (long)(t * BB + m) * G4H;
        float gi = acc[0][r] + bf2f(xrow[0 * HH + jc]);
        float gf = acc[1][r] + bf2f(xrow[1 * HH + jc]);
        float gg = acc[2][r] + bf2f(xrow[2 * HH + jc]);
        float go = acc[3][r] + bf2f(xrow[3 * HH + jc]);
        float co = c_sm[m * HH + jc];
        float cn = sigf(gf) * co + sigf(gi) * tanhf(gg);
        float hn = sigf(go) * tanhf(cn);
        bool mk = t < wordlen[m];
        float ho = bf2f(h_sm[p][m * HH + jc]);
        float h = mk ? hn : ho;
        c_sm[m * HH + jc] = mk ? cn : co;
        h_sm[p ^ 1][m * HH + jc] = f2bf(h);
        // pad_packed zeroes padded outputs
        hs[((long)(m * TT + t) * (2 * HH)) + dir * HH + jc] = mk ? f2bf(h) : (unsigned short)0;
      }
    }
    __syncthreads();
  }
}

// ---------------- k_proj: feats = lstm_out @ projW^T + b  (WMMA) ------------
__global__ __launch_bounds__(256) void k_proj(
    const unsigned short* hs, const unsigned short* projw_bf,
    const float* proj_b, float* feats) {
  int mt = blockIdx.x * 8 + (threadIdx.x >> 5);   // 1024 m-tiles
  int lane = threadIdx.x & 31;
  v8f acc[4];
#pragma unroll
  for (int nt = 0; nt < 4; ++nt) acc[nt] = vzero8();
  const unsigned short* abase = hs + (long)mt * 16 * 512;
  for (int kt = 0; kt < 16; ++kt) {
    v16bf a = load_frag_a(abase + kt * 32, 512, lane);
#pragma unroll
    for (int nt = 0; nt < 4; ++nt) {
      v16bf b = load_frag_b(projw_bf + nt * 16 * 512 + kt * 32, 512, lane);
      acc[nt] = wmma_bf16(a, b, acc[nt]);
    }
  }
#pragma unroll
  for (int nt = 0; nt < 4; ++nt) {
    int n = nt * 16 + (lane & 15);
    float bias = (n < NN) ? proj_b[n] : 0.f;
#pragma unroll
    for (int r = 0; r < 8; ++r) {
      int row = mt * 16 + 8 * (lane >> 4) + r;
      feats[(long)row * 64 + n] = acc[nt][r] + bias;
    }
  }
}

__global__ void k_zero(float* out) { if (blockIdx.x == 0 && threadIdx.x == 0) out[0] = 0.f; }

// ---------------- k_crf: forward Z, gold score, viterbi, backtrace ----------
__global__ __launch_bounds__(64) void k_crf(
    const float* feats, const float* trans, const int* wordlen,
    const int* labels, int* bps, float* out) {
  __shared__ float alpha[NN], vv[NN], tr_s[NN * NN];
  __shared__ int bl;
  int b = blockIdx.x, tid = threadIdx.x;
  int len = wordlen[b];
  for (int i = tid; i < NN * NN; i += 64) tr_s[i] = trans[i];
  const float* fb = feats + (long)b * TT * 64;
  __syncthreads();
  if (tid < NN) {
    alpha[tid] = fb[tid] + tr_s[50 * NN + tid];   // init from START
    vv[tid] = alpha[tid];
  }
  __syncthreads();
  int* bpb = bps + (long)b * TT * NN;
  for (int t = 1; t < TT; ++t) {
    float na = 0.f, nv = 0.f; int nb = tid;
    if (tid < NN) {
      float ft = fb[t * 64 + tid];
      float mx = -1e30f;
      for (int i = 0; i < NN; ++i) mx = fmaxf(mx, alpha[i] + tr_s[i * NN + tid]);
      float se = 0.f;
      for (int i = 0; i < NN; ++i) se += __expf(alpha[i] + tr_s[i * NN + tid] - mx);
      na = mx + __logf(se) + ft;
      float vmx = -1e30f; int am = 0;
      for (int i = 0; i < NN; ++i) {
        float s = vv[i] + tr_s[i * NN + tid];
        if (s > vmx) { vmx = s; am = i; }
      }
      nv = vmx + ft;
      if (t >= len) { na = alpha[tid]; nv = vv[tid]; am = tid; }  // masked + identity bp
      nb = am;
    }
    __syncthreads();
    if (tid < NN) { alpha[tid] = na; vv[tid] = nv; bpb[t * NN + tid] = nb; }
    __syncthreads();
  }
  if (tid == 0) {
    float mx = -1e30f;
    for (int j = 0; j < NN; ++j) mx = fmaxf(mx, alpha[j] + tr_s[j * NN + 51]);
    float se = 0.f;
    for (int j = 0; j < NN; ++j) se += __expf(alpha[j] + tr_s[j * NN + 51] - mx);
    float Z = mx + __logf(se);
    const int* lab = labels + b * TT;
    float gold = tr_s[50 * NN + lab[0]];
    for (int t = 0; t < len; ++t) gold += fb[t * 64 + lab[t]];
    for (int t = 1; t < len; ++t) gold += tr_s[lab[t - 1] * NN + lab[t]];
    gold += tr_s[lab[len - 1] * NN + 51];
    atomicAdd(out, Z - gold);
    float bmx = -1e30f; int bj = 0;
    for (int j = 0; j < NN; ++j) {
      float s = vv[j] + tr_s[j * NN + 51];
      if (s > bmx) { bmx = s; bj = j; }
    }
    bl = bj;
  }
  __syncthreads();
  if (tid == 0) {
    int cur = bl;
    float* tg = out + 1 + (long)b * TT;
    tg[TT - 1] = (TT - 1 < len) ? (float)cur : 0.f;
    for (int t = TT - 2; t >= 0; --t) {
      cur = bpb[(t + 1) * NN + cur];
      tg[t] = (t < len) ? (float)cur : 0.f;
    }
  }
}

// ---------------- host launch ----------------
static inline char* carve(char*& p, size_t bytes) {
  char* r = p;
  p += (bytes + 255) & ~(size_t)255;
  return r;
}

extern "C" void kernel_launch(void* const* d_in, const int* in_sizes, int n_in,
                              void* d_out, int out_size, void* d_ws, size_t ws_size,
                              hipStream_t stream) {
  const int*   p_word   = (const int*)d_in[0];
  const int*   p_feat   = (const int*)d_in[1];
  const int*   p_len    = (const int*)d_in[2];
  const int*   p_char   = (const int*)d_in[3];
  const int*   p_recov  = (const int*)d_in[5];
  const int*   p_label  = (const int*)d_in[7];
  const float* p_cemb   = (const float*)d_in[8];
  const float* p_convw  = (const float*)d_in[9];
  const float* p_convb  = (const float*)d_in[10];
  const float* p_wemb   = (const float*)d_in[11];
  const float* p_femb   = (const float*)d_in[12];
  const float* p_wihf   = (const float*)d_in[13];
  const float* p_whhf   = (const float*)d_in[14];
  const float* p_bf     = (const float*)d_in[15];
  const float* p_wihb   = (const float*)d_in[16];
  const float* p_whhb   = (const float*)d_in[17];
  const float* p_bb     = (const float*)d_in[18];
  const float* p_projw  = (const float*)d_in[19];
  const float* p_projb  = (const float*)d_in[20];
  const float* p_trans  = (const float*)d_in[21];
  float* out = (float*)d_out;

  char* ws = (char*)d_ws;
  unsigned short* wihf_bf  = (unsigned short*)carve(ws, (size_t)G4H * INP * 2);
  unsigned short* wihb_bf  = (unsigned short*)carve(ws, (size_t)G4H * INP * 2);
  unsigned short* whhf_bf  = (unsigned short*)carve(ws, (size_t)G4H * HH * 2);
  unsigned short* whhb_bf  = (unsigned short*)carve(ws, (size_t)G4H * HH * 2);
  unsigned short* projw_bf = (unsigned short*)carve(ws, (size_t)64 * 512 * 2);
  unsigned short* convw_bf = (unsigned short*)carve(ws, (size_t)112 * 160 * 2);
  float*          pooled   = (float*)carve(ws, (size_t)BB * TT * CHH * 4);
  unsigned short* x_bf     = (unsigned short*)carve(ws, (size_t)BB * TT * INP * 2);
  unsigned short* xs_f     = (unsigned short*)carve(ws, (size_t)TT * BB * G4H * 2);
  unsigned short* xs_b     = (unsigned short*)carve(ws, (size_t)TT * BB * G4H * 2);
  unsigned short* hs       = (unsigned short*)carve(ws, (size_t)BB * TT * 512 * 2);
  float*          feats    = (float*)carve(ws, (size_t)BB * TT * 64 * 4);
  int*            bps      = (int*)carve(ws, (size_t)BB * TT * NN * 4);

  k_prep<<<1024, 256, 0, stream>>>(p_wihf, p_wihb, p_whhf, p_whhb, p_projw, p_convw,
                                   wihf_bf, wihb_bf, whhf_bf, whhb_bf, projw_bf, convw_bf);
  k_charcnn<<<BB * TT / 4, 128, 0, stream>>>(p_char, p_cemb, convw_bf, p_convb, pooled);
  k_buildx<<<2048, 256, 0, stream>>>(p_word, p_feat, p_recov, p_wemb, p_femb, pooled, x_bf);
  k_inproj<<<4096, 256, 0, stream>>>(x_bf, wihf_bf, wihb_bf, p_bf, p_bb, xs_f, xs_b);
  k_scan<<<2, 1024, 0, stream>>>(xs_f, xs_b, whhf_bf, whhb_bf, p_len, hs);
  k_proj<<<128, 256, 0, stream>>>(hs, projw_bf, p_projb, feats);
  k_zero<<<1, 1, 0, stream>>>(out);
  k_crf<<<BB, 64, 0, stream>>>(feats, p_trans, p_len, p_label, bps, out);
}
